// Model_1700807050028
// MI455X (gfx1250) — compile-verified, matrix operand order checked by reading, not run on hardware
//
#include <hip/hip_runtime.h>

typedef __attribute__((ext_vector_type(16))) _Float16 v16h;
typedef __attribute__((ext_vector_type(8)))  float    v8f;
typedef __attribute__((ext_vector_type(4)))  unsigned int u32x4;
typedef __attribute__((ext_vector_type(8)))  int      i32x8;
typedef __attribute__((ext_vector_type(4)))  int      i32x4;

#define DEVINL __device__ __forceinline__

#if defined(__has_builtin)
#if __has_builtin(__builtin_amdgcn_tensor_load_to_lds) && \
    __has_builtin(__builtin_amdgcn_s_wait_tensorcnt)
#define USE_TDM 1
#endif
#endif

constexpr int Bc = 128, Wc = 64, Mc = 512, Hc = 64, Cc = 4, LRc = 16, RDc = 32, Dc = 128;
constexpr int G3H = 3 * Hc;              // 192
constexpr int ROWS_BM = Bc * Mc;         // 65536
constexpr int ROWS_BD = Bc * Dc;         // 16384

// ---- WMMA fragment helpers (wave32, 16x16x32 f16 -> f32) -------------------
// A: row-major [rows][K]; B packed as [N][K] so both operands load as two
// contiguous 128-bit reads per lane.
// 16-bit A/B map: lanes 0-15 -> K = {k0..k0+7, k0+16..k0+23}, lanes 16-31 -> +8.
// C/D: lane<16 -> row i, lane>=16 -> row i+8; col = lane%16.
DEVINL v16h load_frag(const _Float16* p, int ld, int major0, int k0) {
  int lane = threadIdx.x & 31;
  int lr = lane & 15, hf = lane >> 4;
  const _Float16* base = p + (size_t)(major0 + lr) * ld + (k0 + hf * 8);
  v16h out;
  ((uint4*)&out)[0] = *(const uint4*)(base);
  ((uint4*)&out)[1] = *(const uint4*)(base + 16);
  return out;
}

DEVINL v8f wmma_f16(v16h a, v16h b, v8f c) {
  return __builtin_amdgcn_wmma_f32_16x16x32_f16(false, a, false, b, (short)0, c,
                                                false, false);
}

// fast activations: rcp+mul instead of IEEE division sequences
DEVINL float sigf(float x) { return __fdividef(1.0f, 1.0f + __expf(-x)); }
DEVINL float tanhf_fast(float x) {
  return 1.0f - __fdividef(2.0f, __expf(2.0f * x) + 1.0f);
}

// ---- one-time precompute kernels -------------------------------------------

// Gt16[n][m] = major_graph[m][n]  (B^T pack, f16)
__global__ void k_conv_G(const float* __restrict__ G, _Float16* __restrict__ Gt16) {
  int tid = blockIdx.x * 256 + threadIdx.x;           // 512*512
  Gt16[tid] = (_Float16)G[(size_t)(tid & 511) * 512 + (tid >> 9)];
}

// wh0 / wx1 are already [3H][H] = [N][K] for gh = h @ w^T : just convert
__global__ void k_conv_w(const float* __restrict__ wh0, const float* __restrict__ wx1,
                         _Float16* __restrict__ wh0p, _Float16* __restrict__ wx1p) {
  int tid = blockIdx.x * 256 + threadIdx.x;           // 192*64
  wh0p[tid] = (_Float16)wh0[tid];
  wx1p[tid] = (_Float16)wx1[tid];
}

// Aw16[c][l][m] = A_g[c,l] * A_v[c,l,m] / ||A_v[c,l,:]||   ([N=l][K=m] pack)
__global__ void k_wn_A(const float* __restrict__ A_v, const float* __restrict__ A_g,
                       _Float16* __restrict__ Aw16) {
  __shared__ float red[256];
  int cl = blockIdx.x;                                 // 0..C*LR-1
  const float* v = A_v + (size_t)cl * Mc;
  float s = 0.f;
  for (int m = threadIdx.x; m < Mc; m += 256) s += v[m] * v[m];
  red[threadIdx.x] = s; __syncthreads();
  for (int o = 128; o > 0; o >>= 1) {
    if ((int)threadIdx.x < o) red[threadIdx.x] += red[threadIdx.x + o];
    __syncthreads();
  }
  float scale = A_g[cl] * rsqrtf(red[0]);
  for (int m = threadIdx.x; m < Mc; m += 256)
    Aw16[(size_t)cl * Mc + m] = (_Float16)(v[m] * scale);
}

// Bp16[c][m][l<16] = weight-normed B_v ; l in [16,32) = 0  (K padded to 32)
__global__ void k_wn_B(const float* __restrict__ B_v, const float* __restrict__ B_g,
                       _Float16* __restrict__ Bp16) {
  int tid = blockIdx.x * 256 + threadIdx.x;            // C*M = 2048
  const float* v = B_v + (size_t)tid * LRc;
  float s = 0.f;
  for (int l = 0; l < LRc; ++l) s += v[l] * v[l];
  float scale = B_g[tid] * rsqrtf(s);
  _Float16* dst = Bp16 + (size_t)tid * 32;
  for (int l = 0; l < LRc; ++l) {
    dst[l] = (_Float16)(v[l] * scale);
    dst[16 + l] = (_Float16)0.f;
  }
}

// Cw16[c][r][d] = weight-normed C_v  ([N=r][K=d] pack)
__global__ void k_wn_C(const float* __restrict__ C_v, const float* __restrict__ C_g,
                       _Float16* __restrict__ Cw16) {
  int tid = threadIdx.x;                               // C*RD = 128
  const float* v = C_v + (size_t)tid * Dc;
  float s = 0.f;
  for (int d = 0; d < Dc; ++d) s += v[d] * v[d];
  float scale = C_g[tid] * rsqrtf(s);
  _Float16* dst = Cw16 + (size_t)tid * Dc;
  for (int d = 0; d < Dc; ++d) dst[d] = (_Float16)(v[d] * scale);
}

// gh1c[m][j] = H_init[1,m,:] @ wh1[j,:] + bh1[j]   (constant over t and b)
__global__ void k_gh1(const float* __restrict__ Hinit, const float* __restrict__ wh1,
                      const float* __restrict__ bh1, float* __restrict__ gh1c) {
  int tid = blockIdx.x * 256 + threadIdx.x;            // M*3H
  int m = tid / G3H, j = tid % G3H;
  const float* hv = Hinit + (size_t)(Mc + m) * Hc;     // layer 1
  const float* wv = wh1 + (size_t)j * Hc;
  float s = 0.f;
  for (int h = 0; h < Hc; ++h) s += hv[h] * wv[h];
  gh1c[tid] = s + bh1[j];
}

// h_prev init: broadcast H_init[0] over batch (f32 + f16 copies)
__global__ void k_hinit(const float* __restrict__ Hinit, float* __restrict__ hp32,
                        _Float16* __restrict__ hp16) {
  size_t tid = (size_t)blockIdx.x * 256 + threadIdx.x; // B*M*H
  int h = tid & 63, m = (tid >> 6) & 511;
  float v = Hinit[(size_t)m * Hc + h];
  hp32[tid] = v;
  hp16[tid] = (_Float16)v;
}

// ---- per-step kernels ------------------------------------------------------

// GRU layer 0: gh0 = h_prev @ wh0^T (WMMA), gates fused in epilogue.
// one wave = 16 rows x one h-group (3 gate tiles r/z/n -> 3 accumulators)
__global__ void k_gru0(const _Float16* __restrict__ hp16, const float* __restrict__ hp32,
                       const _Float16* __restrict__ wh0p, const float* __restrict__ x,
                       const float* __restrict__ wx0, const float* __restrict__ bx0,
                       const float* __restrict__ bh0, _Float16* __restrict__ h1_16,
                       _Float16* __restrict__ xp16, int step) {
  int gw = blockIdx.x * 8 + (threadIdx.x >> 5);        // 16384 waves
  int row0 = (gw >> 2) * 16;
  int tt = gw & 3;
  v8f aR = {}, aZ = {}, aN = {};
#pragma unroll
  for (int kk = 0; kk < Hc; kk += 32) {
    v16h af = load_frag(hp16, Hc, row0, kk);
    v16h bR = load_frag(wh0p, Hc, tt * 16, kk);
    v16h bZ = load_frag(wh0p, Hc, 64 + tt * 16, kk);
    v16h bN = load_frag(wh0p, Hc, 128 + tt * 16, kk);
    aR = wmma_f16(af, bR, aR);
    aZ = wmma_f16(af, bZ, aZ);
    aN = wmma_f16(af, bN, aN);
  }
  int lane = threadIdx.x & 31, lr = lane & 15, hf = lane >> 4;
  int h = lr + 16 * tt;
  float bhr = bh0[h], bhz = bh0[64 + h], bhn = bh0[128 + h];
  float wxr = wx0[h], wxz = wx0[64 + h], wxn = wx0[128 + h];
  float bxr = bx0[h], bxz = bx0[64 + h], bxn = bx0[128 + h];
#pragma unroll
  for (int i = 0; i < 8; ++i) {
    int row = row0 + i + hf * 8;
    int b = row >> 9, m = row & (Mc - 1);
    float xt = x[((size_t)b * Wc + step) * Mc + m];
    float r = sigf(xt * wxr + bxr + aR[i] + bhr);
    float z = sigf(xt * wxz + bxz + aZ[i] + bhz);
    float n = tanhf_fast(xt * wxn + bxn + r * (aN[i] + bhn));
    float hp = hp32[(size_t)row * Hc + h];
    float h1 = n + z * (hp - n);
    h1_16[(size_t)row * Hc + h] = (_Float16)h1;
    xp16[((size_t)b * Dc + h) * Mc + m] = (_Float16)h1;   // x_p[b, d=h, m]
  }
}

// GRU layer 1: gx1 = h1 @ wx1^T (WMMA); gh part is the precomputed constant gh1c.
__global__ void k_gru1(const _Float16* __restrict__ h1_16, const _Float16* __restrict__ wx1p,
                       const float* __restrict__ gh1c, const float* __restrict__ bx1,
                       const float* __restrict__ Hinit, _Float16* __restrict__ xp16,
                       float* __restrict__ hp32, _Float16* __restrict__ hp16) {
  int gw = blockIdx.x * 8 + (threadIdx.x >> 5);
  int row0 = (gw >> 2) * 16;
  int tt = gw & 3;
  v8f aR = {}, aZ = {}, aN = {};
#pragma unroll
  for (int kk = 0; kk < Hc; kk += 32) {
    v16h af = load_frag(h1_16, Hc, row0, kk);
    v16h bR = load_frag(wx1p, Hc, tt * 16, kk);
    v16h bZ = load_frag(wx1p, Hc, 64 + tt * 16, kk);
    v16h bN = load_frag(wx1p, Hc, 128 + tt * 16, kk);
    aR = wmma_f16(af, bR, aR);
    aZ = wmma_f16(af, bZ, aZ);
    aN = wmma_f16(af, bN, aN);
  }
  int lane = threadIdx.x & 31, lr = lane & 15, hf = lane >> 4;
  int h = lr + 16 * tt;
  float bxr = bx1[h], bxz = bx1[64 + h], bxn = bx1[128 + h];
#pragma unroll
  for (int i = 0; i < 8; ++i) {
    int row = row0 + i + hf * 8;
    int b = row >> 9, m = row & (Mc - 1);
    const float* g = gh1c + (size_t)m * G3H;
    float r = sigf(aR[i] + bxr + g[h]);
    float z = sigf(aZ[i] + bxz + g[64 + h]);
    float n = tanhf_fast(aN[i] + bxn + r * g[128 + h]);
    float h0v = Hinit[(size_t)(Mc + m) * Hc + h];          // fixed h0[1]
    float h2 = n + z * (h0v - n);
    xp16[((size_t)b * Dc + 64 + h) * Mc + m] = (_Float16)h2;
    hp32[(size_t)row * Hc + h] = h2;                        // next step h_prev
    hp16[(size_t)row * Hc + h] = (_Float16)h2;
  }
}

// x_p0 = relu(x_p @ major_graph): dominant GEMM [16384x512]x[512x512].
// Block = 128-row x 64-col tile; the 64KB B-strip is staged in LDS via the
// Tensor Data Mover (fallback: cooperative b128 copy). Each wave: 4 col tiles.
__global__ void k_mg(const _Float16* __restrict__ xp16, const _Float16* __restrict__ Gt16,
                     float* __restrict__ xp0) {
  __shared__ _Float16 Bs[64 * 512];                    // 64 KB
  int blk = blockIdx.x;                                // 1024 blocks
  int rb = blk >> 3;                                   // 0..127 row blocks
  int cb = blk & 7;                                    // 0..7 col strips
  const _Float16* src = Gt16 + (size_t)cb * 64 * 512;  // contiguous 64KB strip
#ifdef USE_TDM
  if ((threadIdx.x >> 5) == 0) {
    unsigned long long ga = (unsigned long long)src;
    u32x4 g0;
    g0[0] = 1u;                                        // count = 1 descriptor
    g0[1] = 0u;                                        // lds_addr = 0 (Bs)
    g0[2] = (unsigned)ga;                              // global_addr[31:0]
    g0[3] = (unsigned)((ga >> 32) & 0x01ffffffu) | (2u << 30);  // addr[56:32]|type=2
    i32x8 g1;
    g1[0] = (int)(2u << 16);          // data_size = 4B, mask/flags = 0
    g1[1] = (int)(0x4000u << 16);     // tensor_dim0 = 16384 dwords (lo16 @ bits63:48)
    g1[2] = (int)(1u << 16);          // tensor_dim0 hi16 = 0 ; tensor_dim1 = 1
    g1[3] = (int)(0x4000u << 16);     // tensor_dim1 hi = 0 ; tile_dim0 = 16384
    g1[4] = 1;                        // tile_dim1 = 1, tile_dim2 = 0
    g1[5] = 16384;                    // tensor_dim0_stride (48b, lo32)
    g1[6] = 0;
    g1[7] = 0;
    i32x4 gz = {0, 0, 0, 0};
#if __clang_major__ >= 23
    i32x8 gz8 = {0, 0, 0, 0, 0, 0, 0, 0};
    __builtin_amdgcn_tensor_load_to_lds(g0, g1, gz, gz, gz8, 0);
#else
    __builtin_amdgcn_tensor_load_to_lds(g0, g1, gz, gz, 0);
#endif
    __builtin_amdgcn_s_wait_tensorcnt(0);
  }
  __syncthreads();
#else
  for (int idx = threadIdx.x; idx < (64 * 512) / 8; idx += 256)
    ((uint4*)Bs)[idx] = ((const uint4*)src)[idx];
  __syncthreads();
#endif
  int wid = threadIdx.x >> 5;
  int row0 = rb * 128 + wid * 16;
  int lane = threadIdx.x & 31, lr = lane & 15, hf = lane >> 4;
  v8f acc[4] = {};
#pragma unroll 2
  for (int k0 = 0; k0 < Mc; k0 += 32) {
    v16h af = load_frag(xp16, Mc, row0, k0);
#pragma unroll
    for (int j = 0; j < 4; ++j) {
      const _Float16* bb = &Bs[((j * 16 + lr) * 512) + k0 + hf * 8];
      v16h bf;
      ((uint4*)&bf)[0] = *(const uint4*)(bb);
      ((uint4*)&bf)[1] = *(const uint4*)(bb + 16);
      acc[j] = wmma_f16(af, bf, acc[j]);
    }
  }
#pragma unroll
  for (int j = 0; j < 4; ++j) {
#pragma unroll
    for (int i = 0; i < 8; ++i) {
      int row = row0 + i + hf * 8;
      float v = acc[j][i];
      xp0[(size_t)row * Mc + cb * 64 + j * 16 + lr] = v > 0.f ? v : 0.f;
    }
  }
}

// a[b,c,d,l] = x_p . Aw^T + A_b ; one A fragment feeds all 4 c's.
__global__ void k_a(const _Float16* __restrict__ xp16, const _Float16* __restrict__ Aw16,
                    const float* __restrict__ A_b, _Float16* __restrict__ a16) {
  int gw = blockIdx.x * 8 + (threadIdx.x >> 5);        // 1024 waves
  int row0 = gw * 16;
  v8f acc[4] = {};
#pragma unroll 2
  for (int k0 = 0; k0 < Mc; k0 += 32) {
    v16h af = load_frag(xp16, Mc, row0, k0);
#pragma unroll
    for (int c = 0; c < Cc; ++c) {
      v16h bf = load_frag(Aw16 + (size_t)c * LRc * Mc, Mc, 0, k0);
      acc[c] = wmma_f16(af, bf, acc[c]);
    }
  }
  int lane = threadIdx.x & 31, lr = lane & 15, hf = lane >> 4;
#pragma unroll
  for (int c = 0; c < Cc; ++c) {
    float ab = A_b[c * LRc + lr];
#pragma unroll
    for (int i = 0; i < 8; ++i) {
      int row = row0 + i + hf * 8;
      _Float16* dst = a16 + ((size_t)c * ROWS_BD + row) * 32;
      dst[lr] = (_Float16)(acc[c][i] + ab);
      dst[16 + lr] = (_Float16)0.f;
    }
  }
}

// xr = relu(a @ Bw^T + B_b + x_p0), written TRANSPOSED as xrT[c][b][m][d].
// One A fragment feeds 4 column tiles.
__global__ void k_xr(const _Float16* __restrict__ a16, const _Float16* __restrict__ Bp16,
                     const float* __restrict__ B_b, const float* __restrict__ xp0,
                     _Float16* __restrict__ xrT) {
  int gw = blockIdx.x * 8 + (threadIdx.x >> 5);        // C*1024*8 = 32768 waves
  int c = gw >> 13, rem = gw & 8191;
  int rt = rem >> 3, cq = rem & 7;
  int row0 = rt * 16, colb = cq * 64;
  v16h af = load_frag(a16 + (size_t)c * ROWS_BD * 32, 32, row0, 0);
  v8f acc[4] = {};
#pragma unroll
  for (int j = 0; j < 4; ++j) {
    v16h bf = load_frag(Bp16 + (size_t)c * Mc * 32, 32, colb + j * 16, 0);
    acc[j] = wmma_f16(af, bf, acc[j]);
  }
  int lane = threadIdx.x & 31, lr = lane & 15, hf = lane >> 4;
#pragma unroll
  for (int j = 0; j < 4; ++j) {
    int m = colb + j * 16 + lr;
    float bb = B_b[c * Mc + m];
#pragma unroll
    for (int i = 0; i < 8; ++i) {
      int row = row0 + i + hf * 8;                     // row = b*128 + d
      int b = row >> 7, d = row & 127;
      float v = acc[j][i] + bb + xp0[(size_t)row * Mc + m];
      v = v > 0.f ? v : 0.f;
      xrT[(((size_t)c * Bc + b) * Mc + m) * Dc + d] = (_Float16)v;
    }
  }
}

// y = sigmoid(xrT @ Cw^T + C_b); out = sum_r y * FR_w + FR_b (r-reduce fused)
__global__ void k_out(const _Float16* __restrict__ xrT, const _Float16* __restrict__ Cw16,
                      const float* __restrict__ C_b, const float* __restrict__ FR_w,
                      const float* __restrict__ FR_b, float* __restrict__ out, int step) {
  int gw = blockIdx.x * 8 + (threadIdx.x >> 5);
  int c = gw >> 12, rt = gw & 4095;
  int row0 = rt * 16;
  const _Float16* Abase = xrT + (size_t)c * ROWS_BM * Dc;
  const _Float16* Bbase = Cw16 + (size_t)c * RDc * Dc;
  v8f acc0 = {}, acc1 = {};
#pragma unroll
  for (int k0 = 0; k0 < Dc; k0 += 32) {
    v16h af = load_frag(Abase, Dc, row0, k0);
    v16h b0 = load_frag(Bbase, Dc, 0, k0);
    v16h b1 = load_frag(Bbase, Dc, 16, k0);
    acc0 = wmma_f16(af, b0, acc0);
    acc1 = wmma_f16(af, b1, acc1);
  }
  int lane = threadIdx.x & 31, lr = lane & 15, hf = lane >> 4;
  float cb0 = C_b[c * RDc + lr], cb1 = C_b[c * RDc + 16 + lr];
#pragma unroll
  for (int i = 0; i < 8; ++i) {
    int rowg = row0 + i + hf * 8;                      // rowg = b*512 + m
    int b = rowg >> 9, m = rowg & (Mc - 1);
    const float* fw = FR_w + ((size_t)c * Mc + m) * RDc;
    float t = sigf(acc0[i] + cb0) * fw[lr] + sigf(acc1[i] + cb1) * fw[16 + lr];
#pragma unroll
    for (int msk = 1; msk < 16; msk <<= 1) t += __shfl_xor(t, msk, 32);
    if (lr == 0)
      out[(((size_t)c * Bc + b) * Wc + step) * Mc + m] = t + FR_b[c * Mc + m];
  }
}

// ---- host launch -----------------------------------------------------------
extern "C" void kernel_launch(void* const* d_in, const int* in_sizes, int n_in,
                              void* d_out, int out_size, void* d_ws, size_t ws_size,
                              hipStream_t stream) {
  const float* x      = (const float*)d_in[0];
  const float* H_init = (const float*)d_in[1];
  const float* G      = (const float*)d_in[2];
  const float* wx0    = (const float*)d_in[3];
  const float* wh0    = (const float*)d_in[4];
  const float* bx0    = (const float*)d_in[5];
  const float* bh0    = (const float*)d_in[6];
  const float* wx1    = (const float*)d_in[7];
  const float* wh1    = (const float*)d_in[8];
  const float* bx1    = (const float*)d_in[9];
  const float* bh1    = (const float*)d_in[10];
  const float* A_v    = (const float*)d_in[11];
  const float* A_g    = (const float*)d_in[12];
  const float* A_b    = (const float*)d_in[13];
  const float* B_v    = (const float*)d_in[14];
  const float* B_g    = (const float*)d_in[15];
  const float* B_b    = (const float*)d_in[16];
  const float* C_v    = (const float*)d_in[17];
  const float* C_g    = (const float*)d_in[18];
  const float* C_b    = (const float*)d_in[19];
  const float* FR_w   = (const float*)d_in[20];
  const float* FR_b   = (const float*)d_in[21];
  float* out = (float*)d_out;
  (void)in_sizes; (void)n_in; (void)out_size; (void)ws_size;

  char* p = (char*)d_ws;
  auto carve = [&](size_t bytes) {
    char* r = p;
    p += (bytes + 255) & ~(size_t)255;
    return r;
  };
  _Float16* Gt16  = (_Float16*)carve((size_t)Mc * Mc * 2);
  _Float16* Aw16  = (_Float16*)carve((size_t)Cc * LRc * Mc * 2);
  _Float16* Bp16  = (_Float16*)carve((size_t)Cc * Mc * 32 * 2);
  _Float16* Cw16  = (_Float16*)carve((size_t)Cc * RDc * Dc * 2);
  _Float16* wh0p  = (_Float16*)carve((size_t)G3H * Hc * 2);
  _Float16* wx1p  = (_Float16*)carve((size_t)G3H * Hc * 2);
  float*    gh1c  = (float*)   carve((size_t)Mc * G3H * 4);
  float*    hp32  = (float*)   carve((size_t)ROWS_BM * Hc * 4);
  _Float16* hp16  = (_Float16*)carve((size_t)ROWS_BM * Hc * 2);
  _Float16* h1_16 = (_Float16*)carve((size_t)ROWS_BM * Hc * 2);
  _Float16* xp16  = (_Float16*)carve((size_t)Bc * Dc * Mc * 2);
  float*    xp0   = (float*)   carve((size_t)ROWS_BD * Mc * 4);
  _Float16* a16   = (_Float16*)carve((size_t)Cc * ROWS_BD * 32 * 2);
  _Float16* xrT   = (_Float16*)carve((size_t)Cc * ROWS_BM * Dc * 2);

  // one-time packs / precompute
  k_conv_G<<<(Mc * Mc) / 256, 256, 0, stream>>>(G, Gt16);
  k_conv_w<<<(G3H * Hc) / 256, 256, 0, stream>>>(wh0, wx1, wh0p, wx1p);
  k_wn_A<<<Cc * LRc, 256, 0, stream>>>(A_v, A_g, Aw16);
  k_wn_B<<<(Cc * Mc) / 256, 256, 0, stream>>>(B_v, B_g, Bp16);
  k_wn_C<<<1, Cc * RDc, 0, stream>>>(C_v, C_g, Cw16);
  k_gh1<<<(Mc * G3H) / 256, 256, 0, stream>>>(H_init, wh1, bh1, gh1c);
  k_hinit<<<(ROWS_BM * Hc) / 256, 256, 0, stream>>>(H_init, hp32, hp16);

  for (int t = 0; t < Wc; ++t) {
    k_gru0<<<2048, 256, 0, stream>>>(hp16, hp32, wh0p, x, wx0, bx0, bh0, h1_16, xp16, t);
    k_gru1<<<2048, 256, 0, stream>>>(h1_16, wx1p, gh1c, bx1, H_init, xp16, hp32, hp16);
    k_mg<<<1024, 256, 0, stream>>>(xp16, Gt16, xp0);
    k_a<<<128, 256, 0, stream>>>(xp16, Aw16, A_b, a16);
    k_xr<<<4096, 256, 0, stream>>>(a16, Bp16, B_b, xp0, xrT);
    k_out<<<2048, 256, 0, stream>>>(xrT, Cw16, C_b, FR_w, FR_b, out, t);
  }
}
